// AttentionBlock_44547400794201
// MI455X (gfx1250) — matrix operand order repaired and checked by reading, not compiled, not run on hardware
//
#include <hip/hip_runtime.h>

// ---------------- problem dims ----------------
#define SEQ   2048
#define FEAT  1024
#define KD    512
#define VD    512
#define BATCH 8
#define NROWS (BATCH * SEQ)   // 16384
#define NTOT  1536            // q|k|v combined N
#define OUTW  (FEAT + VD)     // 1536

// ---------------- WMMA types ----------------
typedef __bf16 bf16_t;
typedef bf16_t v16bf __attribute__((ext_vector_type(16)));
typedef float  v8f   __attribute__((ext_vector_type(8)));
typedef unsigned int u32x4 __attribute__((ext_vector_type(4)));

union Frag {
    v16bf v;
    u32x4 q[2];
    unsigned short us[16];
};

__device__ __forceinline__ unsigned short f2bf(float f) {
    unsigned u = __float_as_uint(f);
    unsigned r = (u + 0x7FFFu + ((u >> 16) & 1u)) >> 16;
    return (unsigned short)r;
}

__device__ __forceinline__ u32x4 ld16(const unsigned short* p) {
    return *(const u32x4*)p;
}

__device__ __forceinline__ v8f wmma_bf16(v16bf a, v16bf b, v8f c) {
    // (neg_a, A, neg_b, B, c_mod, C, reuse_a, reuse_b)
    return __builtin_amdgcn_wmma_f32_16x16x32_bf16(false, a, false, b,
                                                   (short)0, c, false, false);
}

// ---------------- async global->LDS staging (gfx1250 path w/ fallback) ----------
#if __has_builtin(__builtin_amdgcn_global_load_async_to_lds_b128)
#define HAVE_ASYNC_LDS 1
// builtin parameter pointee type per hipcc diagnostic: int __attribute__((vector_size(16)))
typedef int avec128 __attribute__((vector_size(4 * sizeof(int))));
#endif

__device__ __forceinline__ void stage16(const unsigned short* g, unsigned short* l) {
#ifdef HAVE_ASYNC_LDS
    __builtin_amdgcn_global_load_async_to_lds_b128((avec128*)g, (avec128*)l, 0, 0);
#else
    *(u32x4*)l = ld16(g);
#endif
}

__device__ __forceinline__ void stage_wait() {
#ifdef HAVE_ASYNC_LDS
#if __has_builtin(__builtin_amdgcn_s_wait_asynccnt)
    __builtin_amdgcn_s_wait_asynccnt(0);
#else
    asm volatile("s_wait_asynccnt 0x0" ::: "memory");
#endif
#endif
}

// ---------------- prep: x -> bf16, and copy x into out[:, :1024] ----------------
__global__ void __launch_bounds__(256)
prep_x(const float* __restrict__ x, unsigned short* __restrict__ xb,
       float* __restrict__ out) {
    int i = blockIdx.x * 256 + threadIdx.x;           // 0 .. NROWS*FEAT-1
    float v = x[i];
    xb[i] = f2bf(v);
    int row = i >> 10;            // /FEAT
    int col = i & 1023;           // %FEAT
    out[(long)row * OUTW + col] = v;
}

// ---------------- prep: WT[n][f] = W_sel[f][n%512] as bf16 ----------------
__global__ void __launch_bounds__(256)
prep_wt(const float* __restrict__ Wq, const float* __restrict__ Wk,
        const float* __restrict__ Wv, unsigned short* __restrict__ WT) {
    int i = blockIdx.x * 256 + threadIdx.x;           // 0 .. NTOT*FEAT-1
    int n = i >> 10;              // /FEAT
    int f = i & 1023;             // %FEAT
    int sel = n >> 9;             // 0:q 1:k 2:v
    int nl  = n & 511;
    const float* W = (sel == 0) ? Wq : ((sel == 1) ? Wk : Wv);
    WT[i] = f2bf(W[f * 512 + nl]);
}

// ---------------- fused QKV GEMM: [16384 x 1024] x [1024 x 1536] ----------------
// block = one 16-row M tile (A staged once into LDS, shared by 8 waves) x 8 N-tiles
__global__ void __launch_bounds__(256)
qkv_gemm(const unsigned short* __restrict__ xb, const unsigned short* __restrict__ WT,
         const float* __restrict__ bq, const float* __restrict__ bk,
         const float* __restrict__ bv,
         unsigned short* __restrict__ qo, unsigned short* __restrict__ ko,
         unsigned short* __restrict__ vT) {
    __shared__ unsigned short As[16 * FEAT];          // 32 KB

    int tid  = threadIdx.x;
    int w    = tid >> 5;
    int lane = tid & 31;
    int mt = blockIdx.x / 3;                          // 1024 M tiles
    int nb = blockIdx.x % 3;                          // 3 N-blocks of 8 tiles
    int m0 = mt * 16;
    int nt = nb * 8 + w;
    int n0 = nt * 64;
    int row = lane & 15;
    int hi  = lane >> 4;

    // stage A tile (16 x 1024 bf16) into LDS: 2048 16B chunks, 8 per thread
    {
        const unsigned short* src = xb + (long)m0 * FEAT;
#pragma unroll
        for (int i = 0; i < 8; ++i) {
            int cid = i * 256 + tid;
            stage16(src + cid * 8, As + cid * 8);
        }
        stage_wait();
        __syncthreads();
    }

    const unsigned short* arow = As + row * FEAT + hi * 8;

    v8f acc[4];
#pragma unroll
    for (int t = 0; t < 4; ++t) acc[t] = {};

    for (int ks = 0; ks < 32; ++ks) {
        Frag a;
        a.q[0] = ld16(arow + ks * 32);        // ds_load_b128
        a.q[1] = ld16(arow + ks * 32 + 16);
#pragma unroll
        for (int t = 0; t < 4; ++t) {
            int col = n0 + t * 16 + row;
            const unsigned short* bp = WT + (long)col * FEAT + ks * 32 + hi * 8;
            __builtin_prefetch(bp + 32, 0, 1);        // global_prefetch_b8
            Frag b;
            b.q[0] = ld16(bp);
            b.q[1] = ld16(bp + 16);
            acc[t] = wmma_bf16(a.v, b.v, acc[t]);
        }
    }

    int sel = n0 >> 9;                       // uniform per wave (64 | 512)
    const float* bias = (sel == 0) ? bq : ((sel == 1) ? bk : bv);
    int bidx = m0 / SEQ;                     // batch (tile never straddles)
    int s0   = m0 % SEQ;

#pragma unroll
    for (int t = 0; t < 4; ++t) {
        int col = n0 + t * 16 + row;
        int cl  = col & 511;
        float bb = bias[cl];
        if (sel < 2) {
            unsigned short* dst = (sel == 0) ? qo : ko;
#pragma unroll
            for (int r = 0; r < 8; ++r) {
                int m = m0 + hi * 8 + r;
                dst[(long)m * KD + cl] = f2bf(acc[t][r] + bb);
            }
        } else {
            unsigned int pk[4];
#pragma unroll
            for (int r = 0; r < 4; ++r) {
                unsigned int lo  = f2bf(acc[t][2 * r] + bb);
                unsigned int hi2 = f2bf(acc[t][2 * r + 1] + bb);
                pk[r] = lo | (hi2 << 16);
            }
            u32x4 pv = {pk[0], pk[1], pk[2], pk[3]};
            *(u32x4*)(vT + ((long)(bidx * VD + cl) * SEQ + s0 + hi * 8)) = pv;
        }
    }
}

// ---------------- causal flash attention ----------------
// block = 4 waves sharing (batch, 16-query tile); each wave owns 128 V cols.
// K tile (32 x 512 bf16) staged into LDS per iteration via async copies.
__global__ void __launch_bounds__(128)
attn(const unsigned short* __restrict__ qg, const unsigned short* __restrict__ kg,
     const unsigned short* __restrict__ vT, float* __restrict__ out) {
    __shared__ unsigned short Ks[32 * KD];            // 32 KB, shared by block
    __shared__ unsigned short Pl[4][16 * 32];         // per-wave P staging

    int tid  = threadIdx.x;
    int w    = tid >> 5;
    int lane = tid & 31;
    int wid  = blockIdx.x * 4 + w;
    int b    = wid >> 9;          // 512 waves per batch
    int rem  = wid & 511;
    int qt   = rem >> 2;          // same for all 4 waves in a block
    int vs   = rem & 3;
    int q0   = qt * 16;
    int v0   = vs * 128;
    int row  = lane & 15;
    int hi   = lane >> 4;

    // load Q fragments for the whole K=512 (16 K-steps of 32)
    Frag qf[16];
    const unsigned short* qrow = qg + (long)(b * SEQ + q0 + row) * KD + hi * 8;
#pragma unroll
    for (int ks = 0; ks < 16; ++ks) {
        qf[ks].q[0] = ld16(qrow + ks * 32);
        qf[ks].q[1] = ld16(qrow + ks * 32 + 16);
    }

    v8f O[8];
#pragma unroll
    for (int nt = 0; nt < 8; ++nt) O[nt] = {};
    float mrow[8], lrow[8];
#pragma unroll
    for (int r = 0; r < 8; ++r) { mrow[r] = -1e30f; lrow[r] = 0.f; }

    const float scale = 0.044194173824159216f;   // 1/sqrt(512)
    int ktiles = (q0 + 15) / 32 + 1;             // uniform across block
    unsigned short* Pw = Pl[w];

    for (int kt = 0; kt < ktiles; ++kt) {
        int key0 = kt * 32;

        // stage K tile (32 keys x 512 feats bf16): 2048 chunks, 16 per thread
        {
            const unsigned short* src = kg + (long)(b * SEQ + key0) * KD;
#pragma unroll
            for (int i = 0; i < 16; ++i) {
                int cid = i * 128 + tid;
                stage16(src + cid * 8, Ks + cid * 8);
            }
            stage_wait();
            __syncthreads();
        }

        // S = Q K^T  (16 x 32 logits, 2 n-halves); B frags from LDS
        v8f S[2];
        S[0] = {}; S[1] = {};
#pragma unroll
        for (int h = 0; h < 2; ++h) {
            const unsigned short* krow = Ks + (h * 16 + row) * KD + hi * 8;
#pragma unroll
            for (int ks = 0; ks < 16; ++ks) {
                Frag kf;
                kf.q[0] = ld16(krow + ks * 32);
                kf.q[1] = ld16(krow + ks * 32 + 16);
                S[h] = wmma_bf16(qf[ks].v, kf.v, S[h]);
            }
        }

        // scale + causal mask
#pragma unroll
        for (int h = 0; h < 2; ++h) {
            int key = key0 + h * 16 + row;
#pragma unroll
            for (int r = 0; r < 8; ++r) {
                int qr = q0 + hi * 8 + r;
                float s = S[h][r] * scale;
                S[h][r] = (key <= qr) ? s : -1e30f;
            }
        }

        // online softmax (row reductions across 16-lane halves)
        float pscale[8];
#pragma unroll
        for (int r = 0; r < 8; ++r) {
            float m = fmaxf(S[0][r], S[1][r]);
#pragma unroll
            for (int off = 1; off < 16; off <<= 1)
                m = fmaxf(m, __shfl_xor(m, off, 16));
            float nmax = fmaxf(mrow[r], m);
            pscale[r] = __expf(mrow[r] - nmax);
            mrow[r] = nmax;
            float p0 = __expf(S[0][r] - nmax);
            float p1 = __expf(S[1][r] - nmax);
            S[0][r] = p0; S[1][r] = p1;
            float sum = p0 + p1;
#pragma unroll
            for (int off = 1; off < 16; off <<= 1)
                sum += __shfl_xor(sum, off, 16);
            lrow[r] = lrow[r] * pscale[r] + sum;
        }
#pragma unroll
        for (int nt = 0; nt < 8; ++nt)
#pragma unroll
            for (int r = 0; r < 8; ++r) O[nt][r] *= pscale[r];

        // C-layout -> A-layout for P via per-wave LDS tile (bf16)
#pragma unroll
        for (int h = 0; h < 2; ++h)
#pragma unroll
            for (int r = 0; r < 8; ++r)
                Pw[(hi * 8 + r) * 32 + h * 16 + row] = f2bf(S[h][r]);

        Frag pf;
        {
            const unsigned short* pr = Pw + row * 32 + hi * 8;
            pf.q[0] = ld16(pr);
            pf.q[1] = ld16(pr + 16);
        }

        // O += P V   (8 WMMAs over this wave's 128 V columns)
#pragma unroll
        for (int nt = 0; nt < 8; ++nt) {
            const unsigned short* vrow =
                vT + (long)(b * VD + v0 + nt * 16 + row) * SEQ + key0 + hi * 8;
            Frag vf;
            vf.q[0] = ld16(vrow);
            vf.q[1] = ld16(vrow + 16);
            O[nt] = wmma_bf16(pf.v, vf.v, O[nt]);
        }

        __syncthreads();    // protect Ks before next iteration's staging
    }

    // normalize + write out[:, 1024 + v0 + ...]
#pragma unroll
    for (int r = 0; r < 8; ++r) lrow[r] = 1.0f / lrow[r];
#pragma unroll
    for (int nt = 0; nt < 8; ++nt)
#pragma unroll
        for (int r = 0; r < 8; ++r) {
            int qr = q0 + hi * 8 + r;
            out[(long)(b * SEQ + qr) * OUTW + FEAT + v0 + nt * 16 + row] =
                O[nt][r] * lrow[r];
        }
}

// ---------------- launch ----------------
extern "C" void kernel_launch(void* const* d_in, const int* in_sizes, int n_in,
                              void* d_out, int out_size, void* d_ws, size_t ws_size,
                              hipStream_t stream) {
    (void)in_sizes; (void)n_in; (void)out_size; (void)ws_size;
    const float* x  = (const float*)d_in[0];
    const float* Wq = (const float*)d_in[1];
    const float* bq = (const float*)d_in[2];
    const float* Wk = (const float*)d_in[3];
    const float* bk = (const float*)d_in[4];
    const float* Wv = (const float*)d_in[5];
    const float* bv = (const float*)d_in[6];
    float* out = (float*)d_out;

    char* ws = (char*)d_ws;
    size_t off = 0;
    unsigned short* xb = (unsigned short*)(ws + off); off += (size_t)NROWS * FEAT * 2;
    unsigned short* WT = (unsigned short*)(ws + off); off += (size_t)NTOT * FEAT * 2;
    unsigned short* qo = (unsigned short*)(ws + off); off += (size_t)NROWS * KD * 2;
    unsigned short* ko = (unsigned short*)(ws + off); off += (size_t)NROWS * KD * 2;
    unsigned short* vT = (unsigned short*)(ws + off); off += (size_t)NROWS * VD * 2;

    prep_x<<<(NROWS * FEAT) / 256, 256, 0, stream>>>(x, xb, out);
    prep_wt<<<(NTOT * FEAT) / 256, 256, 0, stream>>>(Wq, Wk, Wv, WT);
    // GEMM: 1024 M-tiles x 3 N-blocks (8 waves of 16x64 tiles each)
    qkv_gemm<<<3072, 256, 0, stream>>>(xb, WT, bq, bk, bv, qo, ko, vT);
    // attention: 8 * 128 qtiles * 4 vslices = 4096 waves / 4 per block
    attn<<<1024, 128, 0, stream>>>(qo, ko, vT, out);
}
